// DeepseekV3Attention_41420664603104
// MI455X (gfx1250) — matrix operand order repaired and checked by reading, not compiled, not run on hardware
//
#include <hip/hip_runtime.h>
#include <hip/hip_bf16.h>
#include <math.h>

// ---- problem constants (fixed by reference) ----
#define B_   2
#define S_   2048
#define HID_ 2048
#define NH_  16
#define QL_  1536
#define KVL_ 512
#define DR_  64
#define DN_  128
#define DV_  128
#define DQK_ 192
#define M_   (B_*S_)          // 4096 tokens

typedef __attribute__((ext_vector_type(16))) __bf16 v16bf;
typedef __attribute__((ext_vector_type(8)))  float  v8f;
typedef __attribute__((ext_vector_type(8)))  unsigned short us8;

static __device__ __forceinline__ unsigned short f2bf(float f){
  unsigned u = __float_as_uint(f);
  u += 0x7FFFu + ((u >> 16) & 1u);     // round-to-nearest-even
  return (unsigned short)(u >> 16);
}
static __device__ __forceinline__ v16bf mk16(us8 lo, us8 hi){
  union { us8 u[2]; v16bf v; } x; x.u[0] = lo; x.u[1] = hi; return x.v;
}
static __device__ __forceinline__ v8f wmma_bf(v16bf a, v16bf b, v8f c){
  return __builtin_amdgcn_wmma_f32_16x16x32_bf16(false, a, false, b, (short)0, c, false, false);
}

// ---------------- fp32 -> bf16 convert ----------------
__global__ void cvt_bf16_kernel(const float* __restrict__ in, unsigned short* __restrict__ out, int n){
  int i = blockIdx.x * blockDim.x + threadIdx.x;
  if (i < n) out[i] = f2bf(in[i]);
}

// ---------------- generic C[M,N] = A[M,K] @ W[N,K]^T  (bf16 in, f32 out) ----------------
// one 16(M) x 64(N) macro-tile per wave: A-operand reused across 4 WMMAs per K-step
// -> 10 b128 loads per 4 wmma (vs 16 for independent tiles)
__global__ void gemm_bf16_kernel(const unsigned short* __restrict__ A,
                                 const unsigned short* __restrict__ W,
                                 float* __restrict__ C, int M, int N, int K){
  int lane = threadIdx.x & 31, wave = threadIdx.x >> 5;
  int tile = blockIdx.x * 8 + wave;
  int ntn = N >> 6;                          // N in units of 64
  int tilesTotal = (M >> 4) * ntn;
  if (tile >= tilesTotal) return;
  int tm = tile / ntn, tn = tile % ntn;
  int half = lane >> 4, l16 = lane & 15;

  const unsigned short* arow = A + (size_t)(tm * 16 + l16) * K;   // A-operand: lane holds row l16
  const unsigned short* wrow0 = W + (size_t)(tn * 64 + l16) * K;  // B-operand: lane holds col l16 of sub-tile j
  const unsigned short* wrow1 = wrow0 + (size_t)16 * K;
  const unsigned short* wrow2 = wrow0 + (size_t)32 * K;
  const unsigned short* wrow3 = wrow0 + (size_t)48 * K;

  v8f acc0 = {}, acc1 = {}, acc2 = {}, acc3 = {};
  for (int kk = 0; kk < K; kk += 32){
    // A 16x32 bf16, k order chosen here (B matches): half 0 -> K 0..15 contiguous, half 1 -> K 16..31
    us8 a0 = *(const us8*)(arow + kk + half * 16);
    us8 a1 = *(const us8*)(arow + kk + half * 16 + 8);
    v16bf av = mk16(a0, a1);
    // B 32x16 bf16, k order chosen here (matches A): half 0 -> K 0..15 contiguous, half 1 -> K 16..31
    us8 b0 = *(const us8*)(wrow0 + kk + half * 16);
    us8 b1 = *(const us8*)(wrow0 + kk + half * 16 + 8);
    acc0 = wmma_bf(av, mk16(b0, b1), acc0);
    us8 b2 = *(const us8*)(wrow1 + kk + half * 16);
    us8 b3 = *(const us8*)(wrow1 + kk + half * 16 + 8);
    acc1 = wmma_bf(av, mk16(b2, b3), acc1);
    us8 b4 = *(const us8*)(wrow2 + kk + half * 16);
    us8 b5 = *(const us8*)(wrow2 + kk + half * 16 + 8);
    acc2 = wmma_bf(av, mk16(b4, b5), acc2);
    us8 b6 = *(const us8*)(wrow3 + kk + half * 16);
    us8 b7 = *(const us8*)(wrow3 + kk + half * 16 + 8);
    acc3 = wmma_bf(av, mk16(b6, b7), acc3);
  }
  // C layout: VGPR r -> row r + 8*half, col = l16 within each 16-wide sub-tile
  float* crow = C + (size_t)(tm * 16 + 8 * half) * N + tn * 64 + l16;
#pragma unroll
  for (int r = 0; r < 8; ++r){
    crow[(size_t)r * N]      = acc0[r];
    crow[(size_t)r * N + 16] = acc1[r];
    crow[(size_t)r * N + 32] = acc2[r];
    crow[(size_t)r * N + 48] = acc3[r];
  }
}

// ---------------- rmsnorm over D, output bf16 ----------------
__global__ void rmsnorm_bf16_kernel(const float* __restrict__ in, int in_stride,
                                    const float* __restrict__ w,
                                    unsigned short* __restrict__ out, int out_stride, int D){
  __shared__ float red[8];
  int row = blockIdx.x;
  const float* x = in + (size_t)row * in_stride;
  float ss = 0.f;
  for (int i = threadIdx.x; i < D; i += 256){ float v = x[i]; ss += v * v; }
#pragma unroll
  for (int off = 16; off; off >>= 1) ss += __shfl_xor(ss, off, 32);
  if ((threadIdx.x & 31) == 0) red[threadIdx.x >> 5] = ss;
  __syncthreads();
  float tot = 0.f;
#pragma unroll
  for (int j = 0; j < 8; ++j) tot += red[j];
  float inv = rsqrtf(tot / (float)D + 1e-6f);
  unsigned short* y = out + (size_t)row * out_stride;
  for (int i = threadIdx.x; i < D; i += 256) y[i] = f2bf(x[i] * inv * w[i]);
}

// ---------------- build Q (rope-interleaved, scaling folded), bf16 [b][h][s][192] ----------------
__global__ void prep_q_kernel(const float* __restrict__ q,
                              const float* __restrict__ cosb, const float* __restrict__ sinb,
                              unsigned short* __restrict__ Qb, float scaling){
  int idx = blockIdx.x;
  int s = idx & (S_ - 1); int h = (idx >> 11) & (NH_ - 1); int b = idx >> 15;
  const float* qrow = q + ((size_t)(b * S_ + s)) * (NH_ * DQK_) + h * DQK_;
  const float* c  = cosb + ((size_t)(b * S_ + s)) * DR_;
  const float* sn = sinb + ((size_t)(b * S_ + s)) * DR_;
  unsigned short* out = Qb + ((size_t)((b * NH_ + h) * S_) + s) * DQK_;
  int t = threadIdx.x;                      // 64 threads
  out[t]      = f2bf(qrow[t]      * scaling);
  out[t + 64] = f2bf(qrow[t + 64] * scaling);
  if (t < 32){
    float x0 = qrow[DN_ + 2 * t], x1 = qrow[DN_ + 2 * t + 1];
    float ci = c[t], si = sn[t];
    out[DN_ + t]      = f2bf((x0 * ci - x1 * si) * scaling);
    out[DN_ + 32 + t] = f2bf((x0 * si + x1 * ci) * scaling);
  }
}

// ---------------- build K (k_pass per head + shared rope part), bf16 [b][h][s][192] ----------------
__global__ void prep_k_kernel(const float* __restrict__ kv, const float* __restrict__ ckv,
                              const float* __restrict__ cosb, const float* __restrict__ sinb,
                              unsigned short* __restrict__ Kb){
  int idx = blockIdx.x;
  int s = idx & (S_ - 1); int h = (idx >> 11) & (NH_ - 1); int b = idx >> 15;
  const float* kvrow = kv + ((size_t)(b * S_ + s)) * (NH_ * (DN_ + DV_)) + h * (DN_ + DV_);
  const float* rot   = ckv + ((size_t)(b * S_ + s)) * (KVL_ + DR_) + KVL_;
  const float* c  = cosb + ((size_t)(b * S_ + s)) * DR_;
  const float* sn = sinb + ((size_t)(b * S_ + s)) * DR_;
  unsigned short* out = Kb + ((size_t)((b * NH_ + h) * S_) + s) * DQK_;
  int t = threadIdx.x;
  out[t]      = f2bf(kvrow[t]);
  out[t + 64] = f2bf(kvrow[t + 64]);
  if (t < 32){
    float x0 = rot[2 * t], x1 = rot[2 * t + 1];
    float ci = c[t], si = sn[t];
    out[DN_ + t]      = f2bf(x0 * ci - x1 * si);
    out[DN_ + 32 + t] = f2bf(x0 * si + x1 * ci);
  }
}

// ---------------- V transpose: Vt[b][h][d][s] bf16 ----------------
__global__ void prep_v_kernel(const float* __restrict__ kv, unsigned short* __restrict__ Vt){
  int i = blockIdx.x * 256 + threadIdx.x;            // total = 2^23
  int s = i & (S_ - 1); int d = (i >> 11) & 127; int h = (i >> 18) & 15; int b = i >> 22;
  Vt[i] = f2bf(kv[((size_t)(b * S_ + s)) * (NH_ * (DN_ + DV_)) + h * (DN_ + DV_) + DN_ + d]);
}

// ---------------- flash attention: 1 wave per 16-row Q tile, causal ----------------
__global__ void mla_attn_kernel(const unsigned short* __restrict__ Qb,
                                const unsigned short* __restrict__ Kb,
                                const unsigned short* __restrict__ Vt,
                                unsigned short* __restrict__ Ob){
  __shared__ __align__(16) unsigned short pbuf[8][16 * 32];
  int lane = threadIdx.x & 31, wave = threadIdx.x >> 5;
  int qt = blockIdx.x * 8 + wave;                    // 0..127
  int h = blockIdx.y, b = blockIdx.z;
  int half = lane >> 4, l16 = lane & 15;
  const unsigned short* Qh = Qb + ((size_t)(b * NH_ + h) * S_) * DQK_;
  const unsigned short* Kh = Kb + ((size_t)(b * NH_ + h) * S_) * DQK_;
  const unsigned short* Vh = Vt + ((size_t)(b * NH_ + h) * DV_) * S_;
  int q0 = qt * 16;

  // preload Q A-operand (16x192 -> 6 chunks of K=32)
  v16bf qa[6];
  const unsigned short* qrow = Qh + (size_t)(q0 + l16) * DQK_;
#pragma unroll
  for (int c = 0; c < 6; ++c){
    us8 a0 = *(const us8*)(qrow + c * 32 + half * 16);
    us8 a1 = *(const us8*)(qrow + c * 32 + half * 16 + 8);
    qa[c] = mk16(a0, a1);
  }

  float m_i[8], l_i[8];
  v8f o[8];
#pragma unroll
  for (int r = 0; r < 8; ++r){ m_i[r] = -1e30f; l_i[r] = 0.f; o[r] = (v8f){}; }

  for (int kb = 0; kb <= q0; kb += 32){              // 32 keys per step
    v8f s0 = {}, s1 = {};
    const unsigned short* krow0 = Kh + (size_t)(kb + l16) * DQK_;
    const unsigned short* krow1 = Kh + (size_t)(kb + 16 + l16) * DQK_;
#pragma unroll
    for (int c = 0; c < 6; ++c){
      us8 b0 = *(const us8*)(krow0 + c * 32 + half * 16);
      us8 b1 = *(const us8*)(krow0 + c * 32 + half * 16 + 8);
      s0 = wmma_bf(qa[c], mk16(b0, b1), s0);
      us8 b2 = *(const us8*)(krow1 + c * 32 + half * 16);
      us8 b3 = *(const us8*)(krow1 + c * 32 + half * 16 + 8);
      s1 = wmma_bf(qa[c], mk16(b2, b3), s1);
    }
    // online softmax per row (row r+8*half lives across the 16 lanes of this half)
#pragma unroll
    for (int r = 0; r < 8; ++r){
      int row = q0 + r + 8 * half;
      float v0 = (kb + l16      <= row) ? s0[r] : -1e30f;
      float v1 = (kb + 16 + l16 <= row) ? s1[r] : -1e30f;
      float mx = fmaxf(v0, v1);
#pragma unroll
      for (int off = 1; off < 16; off <<= 1) mx = fmaxf(mx, __shfl_xor(mx, off, 32));
      float mnew = fmaxf(m_i[r], mx);
      float p0 = __expf(v0 - mnew), p1 = __expf(v1 - mnew);
      float rsum = p0 + p1;
#pragma unroll
      for (int off = 1; off < 16; off <<= 1) rsum += __shfl_xor(rsum, off, 32);
      float corr = __expf(m_i[r] - mnew);
      l_i[r] = l_i[r] * corr + rsum;
      m_i[r] = mnew;
#pragma unroll
      for (int n = 0; n < 8; ++n) o[n][r] *= corr;
      pbuf[wave][(r + 8 * half) * 32 + l16]      = f2bf(p0);
      pbuf[wave][(r + 8 * half) * 32 + 16 + l16] = f2bf(p1);
    }
    // reload P as A-operand (16x32) from wave-private LDS
    const unsigned short* pr = &pbuf[wave][l16 * 32 + half * 16];
    us8 pa0 = *(const us8*)(pr);
    us8 pa1 = *(const us8*)(pr + 8);
    v16bf pa = mk16(pa0, pa1);
    // O += P @ V  (8 chunks of 16 v-dims; V^T rows are contiguous over keys)
#pragma unroll
    for (int n = 0; n < 8; ++n){
      const unsigned short* vr = Vh + (size_t)(n * 16 + l16) * S_ + kb + half * 16;
      us8 b0 = *(const us8*)(vr);
      us8 b1 = *(const us8*)(vr + 8);
      o[n] = wmma_bf(pa, mk16(b0, b1), o[n]);
    }
  }
  // epilogue: normalize, write attn [b][s][h*128+d] as bf16 for the Wo GEMM
#pragma unroll
  for (int r = 0; r < 8; ++r){
    float inv = 1.0f / l_i[r];
    int row = q0 + r + 8 * half;
    unsigned short* orow = Ob + ((size_t)(b * S_ + row)) * (NH_ * DV_) + h * DV_;
#pragma unroll
    for (int n = 0; n < 8; ++n) orow[n * 16 + l16] = f2bf(o[n][r] * inv);
  }
}

// ---------------- host orchestration ----------------
extern "C" void kernel_launch(void* const* d_in, const int* in_sizes, int n_in,
                              void* d_out, int out_size, void* d_ws, size_t ws_size,
                              hipStream_t stream) {
  const float* hidden = (const float*)d_in[0];
  const float* cosb   = (const float*)d_in[1];
  const float* sinb   = (const float*)d_in[2];
  /* d_in[3] = attention_mask: causal, handled analytically */
  const float* Wqa    = (const float*)d_in[4];
  const float* q_ln   = (const float*)d_in[5];
  const float* Wqb    = (const float*)d_in[6];
  const float* Wkva   = (const float*)d_in[7];
  const float* kv_ln  = (const float*)d_in[8];
  const float* Wkvb   = (const float*)d_in[9];
  const float* Wo     = (const float*)d_in[10];

  char* p = (char*)d_ws;
  auto alloc = [&](size_t bytes)->void*{ void* r = (void*)p; p += (bytes + 255) & ~(size_t)255; return r; };

  unsigned short* hs_bf   = (unsigned short*)alloc((size_t)M_ * HID_ * 2);
  unsigned short* Wqa_bf  = (unsigned short*)alloc((size_t)QL_ * HID_ * 2);
  unsigned short* Wqb_bf  = (unsigned short*)alloc((size_t)NH_ * DQK_ * QL_ * 2);
  unsigned short* Wkva_bf = (unsigned short*)alloc((size_t)(KVL_ + DR_) * HID_ * 2);
  unsigned short* Wkvb_bf = (unsigned short*)alloc((size_t)NH_ * (DN_ + DV_) * KVL_ * 2);
  unsigned short* Wo_bf   = (unsigned short*)alloc((size_t)HID_ * NH_ * DV_ * 2);
  float*          qa_f    = (float*)alloc((size_t)M_ * QL_ * 4);
  unsigned short* qan_bf  = (unsigned short*)alloc((size_t)M_ * QL_ * 2);
  float*          q_f     = (float*)alloc((size_t)M_ * NH_ * DQK_ * 4);
  float*          ckv_f   = (float*)alloc((size_t)M_ * (KVL_ + DR_) * 4);
  unsigned short* kvn_bf  = (unsigned short*)alloc((size_t)M_ * KVL_ * 2);
  float*          kv_f    = (float*)alloc((size_t)M_ * NH_ * (DN_ + DV_) * 4);
  unsigned short* Qb      = (unsigned short*)alloc((size_t)B_ * NH_ * S_ * DQK_ * 2);
  unsigned short* Kb      = (unsigned short*)alloc((size_t)B_ * NH_ * S_ * DQK_ * 2);
  unsigned short* Vt      = (unsigned short*)alloc((size_t)B_ * NH_ * DV_ * S_ * 2);
  unsigned short* attn_bf = (unsigned short*)alloc((size_t)M_ * NH_ * DV_ * 2);

  auto cvt = [&](const float* in, unsigned short* out, int n){
    hipLaunchKernelGGL(cvt_bf16_kernel, dim3((n + 255) / 256), dim3(256), 0, stream, in, out, n);
  };
  auto gemm = [&](const unsigned short* A, const unsigned short* W, float* C, int M, int N, int K){
    int tiles = (M >> 4) * (N >> 6);     // 16x64 macro-tiles, 8 waves/block
    hipLaunchKernelGGL(gemm_bf16_kernel, dim3((tiles + 7) / 8), dim3(256), 0, stream, A, W, C, M, N, K);
  };

  // 1) fp32 -> bf16 staging
  cvt(hidden, hs_bf,  M_ * HID_);
  cvt(Wqa,  Wqa_bf,  QL_ * HID_);
  cvt(Wqb,  Wqb_bf,  NH_ * DQK_ * QL_);
  cvt(Wkva, Wkva_bf, (KVL_ + DR_) * HID_);
  cvt(Wkvb, Wkvb_bf, NH_ * (DN_ + DV_) * KVL_);
  cvt(Wo,   Wo_bf,   HID_ * NH_ * DV_);

  // 2) low-rank projections
  gemm(hs_bf, Wqa_bf,  qa_f,  M_, QL_,        HID_);
  gemm(hs_bf, Wkva_bf, ckv_f, M_, KVL_ + DR_, HID_);

  // 3) rmsnorm (output bf16)
  hipLaunchKernelGGL(rmsnorm_bf16_kernel, dim3(M_), dim3(256), 0, stream,
                     qa_f, QL_, q_ln, qan_bf, QL_, QL_);
  hipLaunchKernelGGL(rmsnorm_bf16_kernel, dim3(M_), dim3(256), 0, stream,
                     ckv_f, KVL_ + DR_, kv_ln, kvn_bf, KVL_, KVL_);

  // 4) up projections
  gemm(qan_bf, Wqb_bf,  q_f,  M_, NH_ * DQK_,        QL_);
  gemm(kvn_bf, Wkvb_bf, kv_f, M_, NH_ * (DN_ + DV_), KVL_);

  // 5) RoPE + head layout + V transpose
  double ms = 0.1 * log(40.0) + 1.0;
  float scaling = (float)(ms * ms / sqrt((double)DQK_));
  hipLaunchKernelGGL(prep_q_kernel, dim3(B_ * NH_ * S_), dim3(64), 0, stream, q_f, cosb, sinb, Qb, scaling);
  hipLaunchKernelGGL(prep_k_kernel, dim3(B_ * NH_ * S_), dim3(64), 0, stream, kv_f, ckv_f, cosb, sinb, Kb);
  hipLaunchKernelGGL(prep_v_kernel, dim3((B_ * NH_ * DV_ * S_) / 256), dim3(256), 0, stream, kv_f, Vt);

  // 6) causal flash attention (WMMA for QK^T and PV)
  hipLaunchKernelGGL(mla_attn_kernel, dim3(S_ / 16 / 8, NH_, B_), dim3(256), 0, stream, Qb, Kb, Vt, attn_bf);

  // 7) output projection into f32 d_out
  gemm(attn_bf, Wo_bf, (float*)d_out, M_, HID_, NH_ * DV_);
}
